// MultiHeadAttention_17179869871
// MI455X (gfx1250) — matrix operand order repaired and checked by reading, not compiled, not run on hardware
//
#include <hip/hip_runtime.h>
#include <stdint.h>

// ---------------------------------------------------------------------------
// MHA for MI455X (gfx1250, wave32, WMMA + Tensor Data Mover).
// bf16 matrix inputs, f32 accumulate.
// ---------------------------------------------------------------------------

#define DMODEL 1024
#define TSEQ   2048
#define BATCH  4
#define HEADS  16
#define DK     64
#define MTOT   (BATCH * TSEQ)   // 8192 rows

typedef __attribute__((ext_vector_type(16))) __bf16 v16bf;
typedef __attribute__((ext_vector_type(8)))  float  v8f;
typedef __attribute__((ext_vector_type(4)))  unsigned int v4u;
typedef __attribute__((ext_vector_type(8)))  int    v8i;
typedef __attribute__((ext_vector_type(4)))  int    v4i;

#ifndef __has_builtin
#define __has_builtin(x) 0
#endif
#if __has_builtin(__builtin_amdgcn_tensor_load_to_lds) && \
    __has_builtin(__builtin_amdgcn_s_wait_tensorcnt)
#define HAVE_TDM 1
#else
#define HAVE_TDM 0
#endif

__device__ __forceinline__ uint16_t f32_to_bf16(float f) {
  uint32_t u = __builtin_bit_cast(uint32_t, f);
  uint32_t r = u + 0x7fffu + ((u >> 16) & 1u);   // round-to-nearest-even
  return (uint16_t)(r >> 16);
}

__device__ __forceinline__ v8f wmma_bf16(v16bf a, v16bf b, v8f c) {
  // D = A(16x32 bf16) x B(32x16 bf16) + C(16x16 f32)
  return __builtin_amdgcn_wmma_f32_16x16x32_bf16(
      /*neg_a=*/false, a, /*neg_b=*/false, b,
      /*c_mod=*/(short)0, c, /*reuse_a=*/false, /*reuse_b=*/false);
}

// A-fragment (16x32, MxK): lane m = lane&15; per-lane 16 contiguous
// k-elements at offset 16*(lane>=16).
__device__ __forceinline__ v16bf frag_a(const uint16_t* rowbase, int lane_hi) {
  const __bf16* p = reinterpret_cast<const __bf16*>(rowbase);
  const int o = lane_hi ? 16 : 0;
  v16bf f;
#pragma unroll
  for (int i = 0; i < 16; ++i) f[i] = p[o + i];
  return f;
}

// B-fragment (32x16, KxN) from a row-major (n, k) tile: lane n = lane&15,
// per-lane 16 contiguous k-halves at offset 16*(lane>=16).
__device__ __forceinline__ v16bf frag_b(const uint16_t* rowbase, int lane_hi) {
  const __bf16* p = reinterpret_cast<const __bf16*>(rowbase) + (lane_hi ? 16 : 0);
  v16bf f;
#pragma unroll
  for (int i = 0; i < 16; ++i) f[i] = p[i];
  return f;
}

#if HAVE_TDM
// LDS aperture flat address keeps the workgroup-relative byte offset in [31:0].
__device__ __forceinline__ uint32_t lds_offset(const void* p) {
  return (uint32_t)(uintptr_t)p;
}

// Issue one TDM 2D tile load: data_size = 2 bytes, tile (tile_d0 x tile_d1),
// source row stride stride_d0 (elements). tile_d1 == 0 -> 1D contiguous tile.
__device__ __forceinline__ void tdm_load_2d(uint32_t lds_addr, const void* gaddr,
                                            uint32_t tile_d0, uint32_t tile_d1,
                                            uint32_t tensor_d0, uint32_t tensor_d1,
                                            uint64_t stride_d0) {
  const uint64_t ga = (uint64_t)(uintptr_t)gaddr;
  v4u g0;
  g0[0] = 1u;                                               // count=1 (user D#)
  g0[1] = lds_addr;                                         // LDS byte address
  g0[2] = (uint32_t)ga;                                     // global addr [31:0]
  g0[3] = (uint32_t)((ga >> 32) & 0x01FFFFFFu) | (2u << 30); // addr[56:32] | type=2
  v8i g1;
  g1[0] = (int)(1u << 16);                                  // data_size=1 -> 2 bytes
  g1[1] = (int)((tensor_d0 & 0xFFFFu) << 16);               // tensor_dim0[15:0]
  g1[2] = (int)((tensor_d0 >> 16) | ((tensor_d1 & 0xFFFFu) << 16));
  g1[3] = (int)((tensor_d1 >> 16) | (tile_d0 << 16));       // tile_dim0
  g1[4] = (int)(tile_d1 & 0xFFFFu);                         // tile_dim1 (tile_dim2=0)
  g1[5] = (int)(uint32_t)(stride_d0 & 0xFFFFFFFFu);         // dim0 stride low 32
  g1[6] = (int)(uint32_t)((stride_d0 >> 32) & 0xFFFFu);     // stride high 16
  g1[7] = 0;
  const v4i z = {0, 0, 0, 0};
#if defined(__clang_major__) && (__clang_major__ >= 23)
  const v8i z8 = {0, 0, 0, 0, 0, 0, 0, 0};
  __builtin_amdgcn_tensor_load_to_lds(g0, g1, z, z, z8, 0);
#else
  __builtin_amdgcn_tensor_load_to_lds(g0, g1, z, z, 0);
#endif
}
#endif  // HAVE_TDM

// ---------------------------------------------------------------------------
// GEMM: C[8192 x 1024] = A[8192 x 1024] @ W^T[1024 x 1024] + bias
// W stored torch-style (out, in) row-major == B-tile (n, k) row-major.
// MODE 0: -> Q  (b,h,t,d) bf16         MODE 1: -> K  (b,h,t,d) bf16
// MODE 2: -> V^T (b,h,d,t) bf16        MODE 3: A is bf16 ctx -> f32 out
// Block tile 128x128x32, 256 threads = 8 waves (4x2), wave tile 32x64.
// ---------------------------------------------------------------------------
template <int MODE>
__global__ __launch_bounds__(256) void gemm_proj(const void* __restrict__ Aptr,
                                                 const float* __restrict__ W,
                                                 const float* __restrict__ bias,
                                                 void* __restrict__ Out) {
  __shared__ __align__(16) uint16_t lx[128 * 32];
  __shared__ __align__(16) uint16_t lw[128 * 32];

  const int tid = threadIdx.x;
  const int lane = tid & 31;
  const int wave = tid >> 5;
  const int lane_hi = lane >> 4;
  const int lo = lane & 15;
  const int m0 = blockIdx.x * 128;
  const int n0 = blockIdx.y * 128;
  const int wm = (wave & 3) * 32;   // wave row offset in tile
  const int wn = (wave >> 2) * 64;  // wave col offset in tile

  const float*    Af = (const float*)Aptr;
  const uint16_t* Ah = (const uint16_t*)Aptr;

  v8f acc[2][4];
#pragma unroll
  for (int s = 0; s < 2; ++s)
#pragma unroll
    for (int j = 0; j < 4; ++j)
#pragma unroll
      for (int r = 0; r < 8; ++r) acc[s][j][r] = 0.0f;

  for (int k0 = 0; k0 < DMODEL; k0 += 32) {
    // Cooperative tile fill (fp32 -> bf16 convert on the way into LDS).
#pragma unroll 4
    for (int e = tid; e < 128 * 32; e += 256) {
      const int r = e >> 5, c = e & 31;
      if constexpr (MODE == 3) {
        lx[e] = Ah[(size_t)(m0 + r) * DMODEL + k0 + c];
      } else {
        lx[e] = f32_to_bf16(Af[(size_t)(m0 + r) * DMODEL + k0 + c]);
      }
      lw[e] = f32_to_bf16(W[(size_t)(n0 + r) * DMODEL + k0 + c]);
    }
    // Prefetch next k-tile toward L2/WGP$ (global_prefetch_b8).
    if (k0 + 32 < DMODEL) {
      const int pr = tid >> 1;           // 128 rows, 2 threads per row
      const int pc = (tid & 1) * 16;
      if constexpr (MODE == 3)
        __builtin_prefetch(&Ah[(size_t)(m0 + pr) * DMODEL + k0 + 32 + pc], 0, 1);
      else
        __builtin_prefetch(&Af[(size_t)(m0 + pr) * DMODEL + k0 + 32 + pc], 0, 1);
      __builtin_prefetch(&W[(size_t)(n0 + pr) * DMODEL + k0 + 32 + pc], 0, 1);
    }
    __syncthreads();

    v16bf a[2], b[4];
#pragma unroll
    for (int s = 0; s < 2; ++s)
      a[s] = frag_a(&lx[(wm + s * 16 + lo) * 32], lane_hi);
#pragma unroll
    for (int j = 0; j < 4; ++j)
      b[j] = frag_b(&lw[(wn + j * 16 + lo) * 32], lane_hi);
#pragma unroll
    for (int s = 0; s < 2; ++s)
#pragma unroll
      for (int j = 0; j < 4; ++j)
        acc[s][j] = wmma_bf16(a[s], b[j], acc[s][j]);
    __syncthreads();
  }

  // Epilogue: C-layout element (M = r + 8*lane_hi, N = lane&15) per 16x16 tile.
#pragma unroll
  for (int s = 0; s < 2; ++s) {
#pragma unroll
    for (int j = 0; j < 4; ++j) {
#pragma unroll
      for (int r = 0; r < 8; ++r) {
        const int m = m0 + wm + s * 16 + r + 8 * lane_hi;
        const int n = n0 + wn + j * 16 + lo;
        const float v = acc[s][j][r] + bias[n];
        if constexpr (MODE == 3) {
          ((float*)Out)[(size_t)m * DMODEL + n] = v;
        } else {
          const int bb = m >> 11, t = m & (TSEQ - 1);
          const int h = n >> 6, d = n & (DK - 1);
          const uint16_t hv = f32_to_bf16(v);
          if constexpr (MODE == 2)  // V transposed: (b,h,d,t)
            ((uint16_t*)Out)[(((size_t)(bb * HEADS + h) * DK + d) * TSEQ) + t] = hv;
          else                       // Q/K: (b,h,t,d)
            ((uint16_t*)Out)[(((size_t)(bb * HEADS + h) * TSEQ + t) * DK) + d] = hv;
        }
      }
    }
  }
}

// ---------------------------------------------------------------------------
// Flash attention, causal. Grid (B*H, T/64); 128 threads = 4 waves;
// each wave owns 16 query rows. Per 32-key chunk:
//   TDM double-buffered K/V staging -> QK^T (4 WMMA) -> online softmax
//   (shfl_xor within 16-lane row groups) -> P via LDS -> PV (4 WMMA).
// Q,K: (b,h,t,d) bf16.  Vt: (b,h,d,t) bf16.  Ctx out: (b,t,h*d) bf16.
// ---------------------------------------------------------------------------
__global__ __launch_bounds__(128) void flash_attn(const uint16_t* __restrict__ Q,
                                                  const uint16_t* __restrict__ K,
                                                  const uint16_t* __restrict__ Vt,
                                                  uint16_t* __restrict__ Ctx) {
  __shared__ __align__(16) uint16_t ldsK[2][32 * DK];   // (key, d)   chunk, 4 KB each
  __shared__ __align__(16) uint16_t ldsV[2][DK * 32];   // (d, key)   chunk, 4 KB each
  __shared__ __align__(16) uint16_t lp[4][16 * 32];     // per-wave P tiles

  const int tid = threadIdx.x;
  const int lane = tid & 31;
  const int wave = tid >> 5;
  const int lane_hi = lane >> 4;
  const int lo = lane & 15;
  const int bh = blockIdx.x;             // b*HEADS + h
  const int qt = blockIdx.y;             // 64-row query tile
  const int q0 = qt * 64 + wave * 16;    // wave's first query row

  const uint16_t* Qb = Q  + (size_t)bh * TSEQ * DK;
  const uint16_t* Kb = K  + (size_t)bh * TSEQ * DK;
  const uint16_t* Vb = Vt + (size_t)bh * DK * TSEQ;

  // Q fragments: two 32-wide k-steps over d_k=64. Loaded once, from global.
  v16bf qa[2];
  qa[0] = frag_a(Qb + (size_t)(q0 + lo) * DK + 0,  lane_hi);
  qa[1] = frag_a(Qb + (size_t)(q0 + lo) * DK + 32, lane_hi);

  v8f o[4];
#pragma unroll
  for (int j = 0; j < 4; ++j)
#pragma unroll
    for (int r = 0; r < 8; ++r) o[j][r] = 0.0f;

  float mrow[8], lrow[8];
#pragma unroll
  for (int r = 0; r < 8; ++r) { mrow[r] = -1.0e30f; lrow[r] = 0.0f; }

  const float scale = 0.125f;               // 1/sqrt(64)
  const float NEG_INF = -__builtin_inff();
  const int kend = qt * 64 + 64;            // causal upper bound for block

#if HAVE_TDM
  // Prologue: DMA chunk 0 into buffer 0 (K chunk is one contiguous 4 KB run;
  // V chunk is a 32x64 2D tile with 2048-element row stride).
  if (wave == 0) {
    tdm_load_2d(lds_offset(&ldsK[0][0]), Kb, /*tile*/ 32 * DK, 0,
                /*tensor*/ 1u << 24, 1, 32 * DK);
    tdm_load_2d(lds_offset(&ldsV[0][0]), Vb, /*tile*/ 32, DK,
                /*tensor*/ TSEQ, DK, TSEQ);
  }
#endif

  int cur = 0;
  for (int kb = 0; kb < kend; kb += 32, cur ^= 1) {
#if HAVE_TDM
    // Double-buffer: issue next chunk's DMA, then retire the current chunk.
    // Each chunk = 2 descriptors; TENSORcnt retires in order per wave.
    if (wave == 0) {
      if (kb + 32 < kend) {
        tdm_load_2d(lds_offset(&ldsK[cur ^ 1][0]),
                    Kb + (size_t)(kb + 32) * DK, 32 * DK, 0, 1u << 24, 1, 32 * DK);
        tdm_load_2d(lds_offset(&ldsV[cur ^ 1][0]),
                    Vb + (kb + 32), 32, DK, TSEQ, DK, TSEQ);
        __builtin_amdgcn_s_wait_tensorcnt(2);
      } else {
        __builtin_amdgcn_s_wait_tensorcnt(0);
      }
    }
    __syncthreads();
#else
    // Fallback staging: cooperative copy by all 128 threads.
#pragma unroll 4
    for (int e = tid; e < 32 * DK; e += 128)
      ldsK[cur][e] = Kb[(size_t)kb * DK + e];
#pragma unroll 4
    for (int e = tid; e < DK * 32; e += 128) {
      const int dr = e >> 5, c = e & 31;
      ldsV[cur][e] = Vb[(size_t)dr * TSEQ + kb + c];
    }
    __syncthreads();
#endif

    const uint16_t* Kt = &ldsK[cur][0];
    const uint16_t* Vc = &ldsV[cur][0];

    // ---- S = Q @ K^T for 32 keys (two 16-col tiles) ----
    v8f s0, s1;
#pragma unroll
    for (int r = 0; r < 8; ++r) { s0[r] = 0.0f; s1[r] = 0.0f; }
#pragma unroll
    for (int ks = 0; ks < 2; ++ks) {
      const v16bf kf0 = frag_b(Kt + (size_t)lo * DK + ks * 32, lane_hi);
      const v16bf kf1 = frag_b(Kt + (size_t)(16 + lo) * DK + ks * 32, lane_hi);
      s0 = wmma_bf16(qa[ks], kf0, s0);
      s1 = wmma_bf16(qa[ks], kf1, s1);
    }

    // ---- causal mask + scale ----
#pragma unroll
    for (int r = 0; r < 8; ++r) {
      const int qrow = q0 + r + 8 * lane_hi;
      s0[r] = (kb + lo      <= qrow) ? s0[r] * scale : NEG_INF;
      s1[r] = (kb + 16 + lo <= qrow) ? s1[r] * scale : NEG_INF;
    }

    // ---- online softmax; write P (bf16, A-layout source) to LDS ----
#pragma unroll
    for (int r = 0; r < 8; ++r) {
      float mx = fmaxf(s0[r], s1[r]);
#pragma unroll
      for (int off = 8; off > 0; off >>= 1)
        mx = fmaxf(mx, __shfl_xor(mx, off, 32));
      const float mnew = fmaxf(mrow[r], mx);       // floor -1e30: no NaN
      const float corr = __expf(mrow[r] - mnew);
      const float e0 = __expf(s0[r] - mnew);
      const float e1 = __expf(s1[r] - mnew);
      float rs = e0 + e1;
#pragma unroll
      for (int off = 8; off > 0; off >>= 1)
        rs += __shfl_xor(rs, off, 32);
      lrow[r] = lrow[r] * corr + rs;
      mrow[r] = mnew;
#pragma unroll
      for (int j = 0; j < 4; ++j) o[j][r] *= corr;

      const int row = r + 8 * lane_hi;             // C-layout -> (row, key)
      lp[wave][row * 32 + lo]      = f32_to_bf16(e0);
      lp[wave][row * 32 + 16 + lo] = f32_to_bf16(e1);
    }
    asm volatile("" ::: "memory");   // order LDS P store -> fragment load

    // ---- O += P @ V ----
    const v16bf pa = frag_a(&lp[wave][lo * 32], lane_hi);
#pragma unroll
    for (int j = 0; j < 4; ++j) {
      const v16bf vb = frag_b(Vc + (size_t)(j * 16 + lo) * 32, lane_hi);
      o[j] = wmma_bf16(pa, vb, o[j]);
    }
    __syncthreads();   // everyone done with buffer `cur` before it is re-filled
  }

  // ---- normalize and store context (b, t, h*64+d) bf16 ----
  const int bb = bh >> 4, h = bh & (HEADS - 1);
#pragma unroll
  for (int j = 0; j < 4; ++j) {
#pragma unroll
    for (int r = 0; r < 8; ++r) {
      const int t = q0 + r + 8 * lane_hi;
      const int d = j * 16 + lo;
      const float v = o[j][r] / lrow[r];
      Ctx[((size_t)(bb * TSEQ + t)) * DMODEL + h * DK + d] = f32_to_bf16(v);
    }
  }
}

// ---------------------------------------------------------------------------
// Launch: QKV projections -> flash attention -> output projection.
// Workspace: Q(16MB) | K(16MB) | Vt(16MB) | Ctx(16MB), all bf16.
// ---------------------------------------------------------------------------
extern "C" void kernel_launch(void* const* d_in, const int* in_sizes, int n_in,
                              void* d_out, int out_size, void* d_ws, size_t ws_size,
                              hipStream_t stream) {
  (void)in_sizes; (void)n_in; (void)out_size; (void)ws_size;
  const float* x  = (const float*)d_in[0];
  // d_in[1] = mask (causal) — computed analytically in-kernel.
  const float* Wq = (const float*)d_in[2];
  const float* bq = (const float*)d_in[3];
  const float* Wk = (const float*)d_in[4];
  const float* bk = (const float*)d_in[5];
  const float* Wv = (const float*)d_in[6];
  const float* bv = (const float*)d_in[7];
  const float* Wo = (const float*)d_in[8];
  const float* bo = (const float*)d_in[9];

  const size_t seg = (size_t)MTOT * DMODEL * sizeof(uint16_t);  // 16 MiB
  uint16_t* Qws = (uint16_t*)d_ws;
  uint16_t* Kws = (uint16_t*)((char*)d_ws + seg);
  uint16_t* Vws = (uint16_t*)((char*)d_ws + 2 * seg);
  uint16_t* Cws = (uint16_t*)((char*)d_ws + 3 * seg);

  const dim3 gg(MTOT / 128, DMODEL / 128);   // (64, 8)
  gemm_proj<0><<<gg, 256, 0, stream>>>(x, Wq, bq, Qws);
  gemm_proj<1><<<gg, 256, 0, stream>>>(x, Wk, bk, Kws);
  gemm_proj<2><<<gg, 256, 0, stream>>>(x, Wv, bv, Vws);

  flash_attn<<<dim3(BATCH * HEADS, TSEQ / 64), 128, 0, stream>>>(Qws, Kws, Vws, Cws);

  gemm_proj<3><<<gg, 256, 0, stream>>>(Cws, Wo, bo, d_out);
}
